// VQ_VAE_72619307040971
// MI455X (gfx1250) — matrix operand order repaired and checked by reading, not compiled, not run on hardware
//
#include <hip/hip_runtime.h>
#include <math.h>

// ---------------- problem constants ----------------
#define B_     16384
#define NE_    512
#define ZD_    32
#define GAMMA_ 0.99f
#define BETA_  1.0f

typedef __bf16 bf16_t;
typedef bf16_t v16bf __attribute__((ext_vector_type(16)));
typedef float  v8f   __attribute__((ext_vector_type(8)));

__device__ __forceinline__ bf16_t f2bf(float f) { return (bf16_t)f; }

// =====================================================================
// WMMA GEMM:  C[M,N] = act( A[M,K] @ W[N,K]^T + bias[N] )
// Requires M % 128 == 0, K % 32 == 0.  N may be ragged (guarded path).
// Block = 256 threads (8 waves).  Wave computes 16(M) x 64(N); block 128x64.
// =====================================================================
__global__ __launch_bounds__(256) void gemm_bf16_wmma(
    const float* __restrict__ A, const float* __restrict__ W,
    const float* __restrict__ bias, float* __restrict__ C,
    int M, int N, int K, int do_relu)
{
  const int wave = threadIdx.x >> 5;
  const int lane = threadIdx.x & 31;
  const int l    = lane & 15;
  const int hi   = lane >> 4;          // 0: low half, 1: high half of wave
  const int m0   = blockIdx.x * 128 + wave * 16;
  const int nb   = blockIdx.y * 64;

  v8f acc[4];
#pragma unroll
  for (int t = 0; t < 4; ++t)
#pragma unroll
    for (int i = 0; i < 8; ++i) acc[t][i] = 0.0f;

  // A fragment rows: lanes 0-15 and 16-31 both cover M rows m0..m0+15;
  // hi selects which K-halves this lane holds (ISA 16-bit A 16x32 layout).
  const float* arow = A + (size_t)(m0 + l) * K + hi * 8;

  if (nb + 64 <= N) {
    // ---------- fast path: all 64 columns valid, no guards ----------
    const float* wrow[4];
#pragma unroll
    for (int t = 0; t < 4; ++t)
      wrow[t] = W + (size_t)(nb + t * 16 + l) * K + hi * 8;

    for (int k0 = 0; k0 < K; k0 += 32) {
      v16bf a;
#pragma unroll
      for (int i = 0; i < 8; ++i) a[i]     = f2bf(arow[k0 + i]);
#pragma unroll
      for (int i = 0; i < 8; ++i) a[8 + i] = f2bf(arow[k0 + 16 + i]);

#pragma unroll
      for (int t = 0; t < 4; ++t) {
        v16bf b;
#pragma unroll
        for (int i = 0; i < 16; ++i) b[i] = f2bf(wrow[t][k0 + i + (i >= 8 ? 8 : 0)]);
        acc[t] = __builtin_amdgcn_wmma_f32_16x16x32_bf16(
            false, a, false, b, (short)0, acc[t], false, false);
      }
    }
  } else {
    // ---------- guarded path: hoist validity out of the k-loop ----------
    const float* wrow[4];
    float mask[4];
#pragma unroll
    for (int t = 0; t < 4; ++t) {
      int col  = nb + t * 16 + l;
      int cc   = col < N ? col : (N - 1);     // clamp to a readable row
      mask[t]  = col < N ? 1.0f : 0.0f;
      wrow[t]  = W + (size_t)cc * K + hi * 8;
    }

    for (int k0 = 0; k0 < K; k0 += 32) {
      v16bf a;
#pragma unroll
      for (int i = 0; i < 8; ++i) a[i]     = f2bf(arow[k0 + i]);
#pragma unroll
      for (int i = 0; i < 8; ++i) a[8 + i] = f2bf(arow[k0 + 16 + i]);

#pragma unroll
      for (int t = 0; t < 4; ++t) {
        v16bf b;
#pragma unroll
        for (int i = 0; i < 16; ++i) b[i] = f2bf(wrow[t][k0 + i + (i >= 8 ? 8 : 0)] * mask[t]);
        acc[t] = __builtin_amdgcn_wmma_f32_16x16x32_bf16(
            false, a, false, b, (short)0, acc[t], false, false);
      }
    }
  }

#pragma unroll
  for (int t = 0; t < 4; ++t) {
    const int col = nb + t * 16 + l;
    if (col < N) {
      const float bv = bias ? bias[col] : 0.0f;
#pragma unroll
      for (int i = 0; i < 8; ++i) {
        float v = acc[t][i] + bv;
        if (do_relu) v = fmaxf(v, 0.0f);
        C[(size_t)(m0 + hi * 8 + i) * N + col] = v;
      }
    }
  }
}

// =====================================================================
// Scalar conv kernels (small channel counts; VALU-bound, L2-friendly)
// =====================================================================

// x[B,1,28,28] -> relu(conv 4x4 s2 p1) -> out[B,16,14,14]
__global__ void conv1_k(const float* __restrict__ x, const float* __restrict__ w,
                        const float* __restrict__ b, float* __restrict__ out)
{
  int idx = blockIdx.x * blockDim.x + threadIdx.x;
  const int total = B_ * 16 * 14 * 14;
  if (idx >= total) return;
  int ox = idx % 14, t = idx / 14;
  int oy = t % 14;   t /= 14;
  int c  = t % 16;
  int n  = t / 16;
  float acc = b[c];
  const float* xi = x + (size_t)n * 784;
  const float* wc = w + c * 16;
#pragma unroll
  for (int ky = 0; ky < 4; ++ky) {
    int iy = oy * 2 - 1 + ky;
    if (iy < 0 || iy >= 28) continue;
#pragma unroll
    for (int kx = 0; kx < 4; ++kx) {
      int ix = ox * 2 - 1 + kx;
      if (ix < 0 || ix >= 28) continue;
      acc += xi[iy * 28 + ix] * wc[ky * 4 + kx];
    }
  }
  out[idx] = fmaxf(acc, 0.0f);
}

// in[B,16,14,14] -> conv 4x4 s2 p1 (+bias, no relu) -> out[B,32,7,7]
__global__ void conv2_k(const float* __restrict__ in, const float* __restrict__ w,
                        const float* __restrict__ b, float* __restrict__ out)
{
  int idx = blockIdx.x * blockDim.x + threadIdx.x;
  const int total = B_ * 32 * 7 * 7;
  if (idx >= total) return;
  int ox = idx % 7, t = idx / 7;
  int oy = t % 7;   t /= 7;
  int c  = t % 32;
  int n  = t / 32;
  float acc = b[c];
  const float* xi = in + (size_t)n * (16 * 196);
  const float* wc = w + c * 256;
  for (int ci = 0; ci < 16; ++ci) {
    const float* xc = xi + ci * 196;
    const float* wk = wc + ci * 16;
#pragma unroll
    for (int ky = 0; ky < 4; ++ky) {
      int iy = oy * 2 - 1 + ky;
      if (iy < 0 || iy >= 14) continue;
#pragma unroll
      for (int kx = 0; kx < 4; ++kx) {
        int ix = ox * 2 - 1 + kx;
        if (ix < 0 || ix >= 14) continue;
        acc += xc[iy * 14 + ix] * wk[ky * 4 + kx];
      }
    }
  }
  out[idx] = acc;
}

// mid = relu(conv3x3_p1(relu(in)))  : in[B,32,7,7], w1[64,32,3,3] -> mid[B,64,7,7]
__global__ void res_conv3_k(const float* __restrict__ in, const float* __restrict__ w1,
                            float* __restrict__ mid)
{
  int idx = blockIdx.x * blockDim.x + threadIdx.x;
  const int total = B_ * 64 * 7 * 7;
  if (idx >= total) return;
  int ox = idx % 7, t = idx / 7;
  int oy = t % 7;   t /= 7;
  int co = t % 64;
  int n  = t / 64;
  float acc = 0.0f;
  const float* xi = in + (size_t)n * (32 * 49);
  const float* wc = w1 + co * (32 * 9);
  for (int ci = 0; ci < 32; ++ci) {
    const float* xc = xi + ci * 49;
    const float* wk = wc + ci * 9;
#pragma unroll
    for (int ky = 0; ky < 3; ++ky) {
      int iy = oy - 1 + ky;
      if (iy < 0 || iy >= 7) continue;
#pragma unroll
      for (int kx = 0; kx < 3; ++kx) {
        int ix = ox - 1 + kx;
        if (ix < 0 || ix >= 7) continue;
        acc += fmaxf(xc[iy * 7 + ix], 0.0f) * wk[ky * 3 + kx];
      }
    }
  }
  mid[idx] = fmaxf(acc, 0.0f);
}

// x = relu(x + conv1x1(mid))  in-place on x[B,32,7,7]; mid[B,64,7,7], w2[32,64]
__global__ void res_fin_k(float* __restrict__ x, const float* __restrict__ mid,
                          const float* __restrict__ w2)
{
  int idx = blockIdx.x * blockDim.x + threadIdx.x;
  const int total = B_ * 32 * 7 * 7;
  if (idx >= total) return;
  int s = idx % 49;
  int c = (idx / 49) % 32;
  int n = idx / (32 * 49);
  const float* mi = mid + (size_t)n * (64 * 49) + s;
  const float* wc = w2 + c * 64;
  float acc = 0.0f;
#pragma unroll 8
  for (int ci = 0; ci < 64; ++ci) acc += mi[ci * 49] * wc[ci];
  x[idx] = fmaxf(x[idx] + acc, 0.0f);
}

// transposed conv 4x4 s2 p1: in[B,32,7,7], w[32,16,4,4] -> relu -> out[B,16,14,14]
__global__ void convt1_k(const float* __restrict__ in, const float* __restrict__ w,
                         const float* __restrict__ b, float* __restrict__ out)
{
  int idx = blockIdx.x * blockDim.x + threadIdx.x;
  const int total = B_ * 16 * 14 * 14;
  if (idx >= total) return;
  int ox = idx % 14, t = idx / 14;
  int oy = t % 14;   t /= 14;
  int co = t % 16;
  int n  = t / 16;
  float acc = b[co];
  const float* xi = in + (size_t)n * (32 * 49);
#pragma unroll
  for (int ky = 0; ky < 4; ++ky) {
    int ny = oy + 1 - ky;
    if (ny < 0 || (ny & 1)) continue;
    int iy = ny >> 1;
    if (iy >= 7) continue;
#pragma unroll
    for (int kx = 0; kx < 4; ++kx) {
      int nx = ox + 1 - kx;
      if (nx < 0 || (nx & 1)) continue;
      int ix = nx >> 1;
      if (ix >= 7) continue;
      for (int ci = 0; ci < 32; ++ci)
        acc += xi[ci * 49 + iy * 7 + ix] * w[((ci * 16 + co) * 4 + ky) * 4 + kx];
    }
  }
  out[idx] = fmaxf(acc, 0.0f);
}

// transposed conv 4x4 s2 p1: in[B,16,14,14], w[16,1,4,4] -> out[B,1,28,28] (no relu)
__global__ void convt2_k(const float* __restrict__ in, const float* __restrict__ w,
                         const float* __restrict__ b, float* __restrict__ out)
{
  int idx = blockIdx.x * blockDim.x + threadIdx.x;
  const int total = B_ * 28 * 28;
  if (idx >= total) return;
  int ox = idx % 28;
  int oy = (idx / 28) % 28;
  int n  = idx / 784;
  float acc = b[0];
  const float* xi = in + (size_t)n * (16 * 196);
#pragma unroll
  for (int ky = 0; ky < 4; ++ky) {
    int ny = oy + 1 - ky;
    if (ny < 0 || (ny & 1)) continue;
    int iy = ny >> 1;
    if (iy >= 14) continue;
#pragma unroll
    for (int kx = 0; kx < 4; ++kx) {
      int nx = ox + 1 - kx;
      if (nx < 0 || (nx & 1)) continue;
      int ix = nx >> 1;
      if (ix >= 14) continue;
      for (int ci = 0; ci < 16; ++ci)
        acc += xi[ci * 196 + iy * 14 + ix] * w[ci * 16 + ky * 4 + kx];
    }
  }
  out[idx] = acc;
}

// =====================================================================
// VQ: one wave per row; each lane scans 16 of the 512 codewords.
// Emits z_q = emb[argmin], and accumulates counts/sums/mse via atomics.
// =====================================================================
__global__ __launch_bounds__(256) void vq_k(
    const float* __restrict__ z, const float* __restrict__ emb,
    float* __restrict__ zq, float* __restrict__ counts,
    float* __restrict__ sums, float* __restrict__ mse_acc)
{
  __shared__ float zs[8][ZD_];
  const int wave = threadIdx.x >> 5;
  const int lane = threadIdx.x & 31;
  const int row  = blockIdx.x * 8 + wave;

  zs[wave][lane] = z[(size_t)row * ZD_ + lane];
  __syncthreads();

  float best = 3.4e38f;
  int   bj   = 0;
  for (int j = lane; j < NE_; j += 32) {
    const float* e = emb + (size_t)j * ZD_;
    float d = 0.0f;
#pragma unroll
    for (int t = 0; t < ZD_; ++t) {
      float df = zs[wave][t] - e[t];
      d += df * df;
    }
    if (d < best) { best = d; bj = j; }
  }
  // wave-wide argmin (wave32), first-index tie-break
  for (int off = 16; off > 0; off >>= 1) {
    float od = __shfl_xor(best, off, 32);
    int   oj = __shfl_xor(bj, off, 32);
    if (od < best || (od == best && oj < bj)) { best = od; bj = oj; }
  }

  const float e  = emb[(size_t)bj * ZD_ + lane];
  zq[(size_t)row * ZD_ + lane] = e;     // straight-through value == z_q
  float df = zs[wave][lane] - e;
  float sq = df * df;
  for (int off = 16; off > 0; off >>= 1) sq += __shfl_xor(sq, off, 32);

  if (lane == 0) {
    atomicAdd(&counts[bj], 1.0f);
    atomicAdd(mse_acc, sq);
  }
  atomicAdd(&sums[(size_t)bj * ZD_ + lane], zs[wave][lane]);
}

// losses + perplexity + EMA codebook update
__global__ __launch_bounds__(512) void vq_fin_k(
    const float* __restrict__ counts, const float* __restrict__ sums,
    const float* __restrict__ n_mat, const float* __restrict__ m_mat,
    const float* __restrict__ mse_acc,
    float* __restrict__ out_losses, float* __restrict__ out_emb)
{
  __shared__ float red[NE_];
  const int j = threadIdx.x;
  const float c  = counts[j];
  const float em = c * (1.0f / (float)B_);
  red[j] = em * logf(em + 1e-10f);
  __syncthreads();
  for (int s = NE_ / 2; s > 0; s >>= 1) {
    if (j < s) red[j] += red[j + s];
    __syncthreads();
  }
  if (j == 0) {
    float mse = mse_acc[0] * (1.0f / ((float)B_ * (float)ZD_));
    out_losses[0] = mse;           // quant_loss
    out_losses[1] = BETA_ * mse;   // commit_loss
    out_losses[2] = expf(-red[0]); // perplexity
  }
  const float nn  = n_mat[j] * GAMMA_ + c * (1.0f - GAMMA_);
  const float inv = 1.0f / nn;
  for (int d = 0; d < ZD_; ++d)
    out_emb[j * ZD_ + d] =
        (m_mat[j * ZD_ + d] * GAMMA_ + sums[j * ZD_ + d] * (1.0f - GAMMA_)) * inv;
}

__global__ void zero_k(float* __restrict__ p, int n)
{
  int i = blockIdx.x * blockDim.x + threadIdx.x;
  if (i < n) p[i] = 0.0f;
}

// =====================================================================
// Orchestration
// =====================================================================
extern "C" void kernel_launch(void* const* d_in, const int* in_sizes, int n_in,
                              void* d_out, int out_size, void* d_ws, size_t ws_size,
                              hipStream_t stream)
{
  const float* x        = (const float*)d_in[0];
  const float* conv1_w  = (const float*)d_in[1];
  const float* conv1_b  = (const float*)d_in[2];
  const float* conv2_w  = (const float*)d_in[3];
  const float* conv2_b  = (const float*)d_in[4];
  const float* res1_w1  = (const float*)d_in[5];
  const float* res1_w2  = (const float*)d_in[6];
  const float* fc1_w    = (const float*)d_in[7];
  const float* fc1_b    = (const float*)d_in[8];
  const float* fc2_w    = (const float*)d_in[9];
  const float* fc2_b    = (const float*)d_in[10];
  const float* emb      = (const float*)d_in[11];
  const float* n_mat    = (const float*)d_in[12];
  const float* m_mat    = (const float*)d_in[13];
  const float* fc3_w    = (const float*)d_in[14];
  const float* fc3_b    = (const float*)d_in[15];
  const float* fc4_w    = (const float*)d_in[16];
  const float* fc4_b    = (const float*)d_in[17];
  const float* res2_w1  = (const float*)d_in[18];
  const float* res2_w2  = (const float*)d_in[19];
  const float* convt1_w = (const float*)d_in[20];
  const float* convt1_b = (const float*)d_in[21];
  const float* convt2_w = (const float*)d_in[22];
  const float* convt2_b = (const float*)d_in[23];

  // workspace layout (floats); two big regions are ping-ponged
  float* ws = (float*)d_ws;
  size_t off = 0;
  float* big1 = ws + off; off += (size_t)B_ * 16 * 196; // h1 / res-mid / convt1-out
  float* big2 = ws + off; off += (size_t)B_ * 32 * 49;  // h2 / res-out(in place) / fc4-out
  float* f1   = ws + off; off += (size_t)B_ * 224;      // fc1-out / fc3-out
  float* z    = ws + off; off += (size_t)B_ * ZD_;
  float* zq   = ws + off; off += (size_t)B_ * ZD_;
  float* counts = ws + off; off += NE_;                 // counts, sums, mse contiguous
  float* sums   = ws + off; off += NE_ * ZD_;
  float* msea   = ws + off; off += 8;
  (void)ws_size; (void)in_sizes; (void)n_in; (void)out_size;

  // output layout: x_rec | quant | commit | perplexity | emb_new
  float* x_rec   = (float*)d_out;
  float* losses  = x_rec + (size_t)B_ * 784;
  float* emb_new = losses + 3;

  const int TB = 256;

  // zero VQ accumulators (graph-replay safe)
  {
    int n = NE_ + NE_ * ZD_ + 8;
    zero_k<<<(n + TB - 1) / TB, TB, 0, stream>>>(counts, n);
  }

  // ---------------- encoder ----------------
  { int n = B_ * 16 * 196; conv1_k<<<(n + TB - 1) / TB, TB, 0, stream>>>(x, conv1_w, conv1_b, big1); }
  { int n = B_ * 32 * 49;  conv2_k<<<(n + TB - 1) / TB, TB, 0, stream>>>(big1, conv2_w, conv2_b, big2); }
  { int n = B_ * 64 * 49;  res_conv3_k<<<(n + TB - 1) / TB, TB, 0, stream>>>(big2, res1_w1, big1); }
  { int n = B_ * 32 * 49;  res_fin_k<<<(n + TB - 1) / TB, TB, 0, stream>>>(big2, big1, res1_w2); }

  // fc1: [B,1568] -> relu -> [B,224]
  { dim3 g(B_ / 128, (224 + 63) / 64);
    gemm_bf16_wmma<<<g, 256, 0, stream>>>(big2, fc1_w, fc1_b, f1, B_, 224, 1568, 1); }
  // fc2: [B,224] -> [B,32]
  { dim3 g(B_ / 128, 1);
    gemm_bf16_wmma<<<g, 256, 0, stream>>>(f1, fc2_w, fc2_b, z, B_, ZD_, 224, 0); }

  // ---------------- VQ ----------------
  vq_k<<<B_ / 8, 256, 0, stream>>>(z, emb, zq, counts, sums, msea);
  vq_fin_k<<<1, NE_, 0, stream>>>(counts, sums, n_mat, m_mat, msea, losses, emb_new);

  // ---------------- decoder ----------------
  // fc3: [B,32] -> relu -> [B,224]
  { dim3 g(B_ / 128, (224 + 63) / 64);
    gemm_bf16_wmma<<<g, 256, 0, stream>>>(zq, fc3_w, fc3_b, f1, B_, 224, ZD_, 1); }
  // fc4: [B,224] -> relu -> [B,1568]  (into big2, viewed as [B,32,7,7])
  { dim3 g(B_ / 128, (1568 + 63) / 64);
    gemm_bf16_wmma<<<g, 256, 0, stream>>>(f1, fc4_w, fc4_b, big2, B_, 1568, 224, 1); }

  { int n = B_ * 64 * 49;  res_conv3_k<<<(n + TB - 1) / TB, TB, 0, stream>>>(big2, res2_w1, big1); }
  { int n = B_ * 32 * 49;  res_fin_k<<<(n + TB - 1) / TB, TB, 0, stream>>>(big2, big1, res2_w2); }
  { int n = B_ * 16 * 196; convt1_k<<<(n + TB - 1) / TB, TB, 0, stream>>>(big2, convt1_w, convt1_b, big1); }
  { int n = B_ * 784;      convt2_k<<<(n + TB - 1) / TB, TB, 0, stream>>>(big1, convt2_w, convt2_b, x_rec); }
}
